// PredictModel_30940944400925
// MI455X (gfx1250) — matrix operand order repaired and checked by reading, not compiled, not run on hardware
//
#include <hip/hip_runtime.h>

#define NUM_CLASSES 80
#define TOPK        200
#define TOPK_PAD    224      // padded to 7 * 32 for WMMA K-steps
#define CONF_T      0.05f
#define NMS_T       0.5f
#define CROPF       300.0f
#define A_ANCH      49104
#define A_PAD       49152    // 48 * 1024
#define BATCH       8
#define NEGV        (-1.0f)
#define PER_TH      48       // A_PAD / 1024 live scores per thread (registers)

typedef __attribute__((ext_vector_type(16))) _Float16 v16h;
typedef __attribute__((ext_vector_type(8)))  float    v8f;

// ---------------------------------------------------------------- init ------
__global__ void k_init(float* __restrict__ out, int n, float* __restrict__ maxbox) {
    int i = blockIdx.x * blockDim.x + threadIdx.x;
    if (i < n) out[i] = 0.0f;
    if (i < BATCH) maxbox[i] = 0.0f;
}

// ------------------------------------------------- decode + score/class -----
// grid = (A_PAD/256, BATCH), block = 256
__global__ void k_decode(const float* __restrict__ conf,
                         const float* __restrict__ reg,
                         const float* __restrict__ anch,
                         float* __restrict__ scores,
                         float* __restrict__ boxes,
                         int*   __restrict__ cls,
                         float* __restrict__ maxbox) {
    const int a = blockIdx.x * blockDim.x + threadIdx.x;
    const int b = blockIdx.y;
    __shared__ float red[256];

    float localmax = 0.0f;
    if (a < A_ANCH) {
        // sigmoid is monotonic: max prob = sigmoid(max logit); argmax on logits
        const float* cp = conf + ((size_t)b * A_ANCH + a) * (NUM_CLASSES + 1);
        float best = cp[1];
        int   bc   = 0;
        for (int c = 1; c < NUM_CLASSES; ++c) {
            float v = cp[c + 1];
            if (v > best) { best = v; bc = c; }   // strict >  => first-max (jnp.argmax)
        }
        float score = 1.0f / (1.0f + expf(-best));

        // decode: anchors (ymin,xmin,ymax,xmax), regressions (dy,dx,dh,dw)
        float a0 = anch[a * 4 + 0], a1 = anch[a * 4 + 1];
        float a2 = anch[a * 4 + 2], a3 = anch[a * 4 + 3];
        float yca = 0.5f * (a0 + a2), xca = 0.5f * (a1 + a3);
        float ha  = a2 - a0,          wa  = a3 - a1;
        const float* rp = reg + ((size_t)b * A_ANCH + a) * 4;
        float w  = expf(rp[3]) * wa;
        float h  = expf(rp[2]) * ha;
        float yc = rp[0] * ha + yca;
        float xc = rp[1] * wa + xca;
        const float inv = 1.0f / CROPF;
        float xmin = fmaxf(xc - 0.5f * w, 0.0f) * inv;
        float ymin = fmaxf(yc - 0.5f * h, 0.0f) * inv;
        float xmax = fminf(xc + 0.5f * w, CROPF - 1.0f) * inv;
        float ymax = fminf(yc + 0.5f * h, CROPF - 1.0f) * inv;

        size_t o = (size_t)b * A_PAD + a;
        scores[o] = score;
        cls[o]    = bc;
        float* bp = boxes + o * 4;
        bp[0] = xmin; bp[1] = ymin; bp[2] = xmax; bp[3] = ymax;

        // true per-image max(boxes) is >= 0 (xmin >= 0 always); clamping negative
        // candidates to 0 preserves the max and keeps uint atomicMax order-exact
        localmax = fmaxf(fmaxf(xmin, ymin), fmaxf(fmaxf(xmax, 0.0f), fmaxf(ymax, 0.0f)));
    }
    red[threadIdx.x] = localmax;
    __syncthreads();
    for (int s = 128; s > 0; s >>= 1) {
        if (threadIdx.x < s) red[threadIdx.x] = fmaxf(red[threadIdx.x], red[threadIdx.x + s]);
        __syncthreads();
    }
    if (threadIdx.x == 0)
        atomicMax((unsigned int*)&maxbox[b], __float_as_uint(red[0]));
}

// ------------------------------------- batched-NMS per-class box offset -----
__global__ void k_offset(const float* __restrict__ boxes,
                         const int*   __restrict__ cls,
                         const float* __restrict__ maxbox,
                         float* __restrict__ boxes_o) {
    const int a = blockIdx.x * blockDim.x + threadIdx.x;
    const int b = blockIdx.y;
    size_t o = (size_t)b * A_PAD + a;
    if (a < A_ANCH) {
        float off = (float)cls[o] * (maxbox[b] + 1.0f);
        boxes_o[o * 4 + 0] = boxes[o * 4 + 0] + off;
        boxes_o[o * 4 + 1] = boxes[o * 4 + 1] + off;
        boxes_o[o * 4 + 2] = boxes[o * 4 + 2] + off;
        boxes_o[o * 4 + 3] = boxes[o * 4 + 3] + off;
    } else {
        boxes_o[o * 4 + 0] = 0.0f; boxes_o[o * 4 + 1] = 0.0f;
        boxes_o[o * 4 + 2] = 0.0f; boxes_o[o * 4 + 3] = 0.0f;
    }
}

// ----------------------------------------------- serial NMS, 1 block/image --
// live scores held in registers (48 per thread); wave32 shuffle argmax;
// only 2 barriers per NMS iteration.
__global__ void __launch_bounds__(1024) k_nms(const float* __restrict__ scores,
                                              const float* __restrict__ boxes_o,
                                              int* __restrict__ idx_k,
                                              int* __restrict__ keep_k) {
    const int b   = blockIdx.x;
    const int tid = threadIdx.x;
    __shared__ float wval[32];
    __shared__ int   widx[32];
    __shared__ float bb[4];
    __shared__ int   bcast_idx;
    __shared__ int   bcast_keep;

    const float* bo = boxes_o + (size_t)b * A_PAD * 4;

    float liv[PER_TH];
#pragma unroll
    for (int i = 0; i < PER_TH; ++i) {
        int a = tid + i * 1024;
        float s = (a < A_ANCH) ? scores[(size_t)b * A_PAD + a] : NEGV;
        liv[i] = (a < A_ANCH && s > CONF_T) ? s : NEGV;
    }
    __builtin_prefetch(bo + (size_t)tid * 4, 0, 1);   // warm L2/WGP$ (global_prefetch_b8)

    for (int t = 0; t < TOPK; ++t) {
        // thread-local argmax (tie -> lowest index, matching jnp.argmax)
        float bv = NEGV - 1.0f;
        int   bi = A_PAD;
#pragma unroll
        for (int i = 0; i < PER_TH; ++i) {
            int a = tid + i * 1024;
            float v = liv[i];
            if (v > bv || (v == bv && a < bi)) { bv = v; bi = a; }
        }
        // wave32 reduction
#pragma unroll
        for (int off = 16; off > 0; off >>= 1) {
            float ov = __shfl_xor(bv, off, 32);
            int   oi = __shfl_xor(bi, off, 32);
            if (ov > bv || (ov == bv && oi < bi)) { bv = ov; bi = oi; }
        }
        if ((tid & 31) == 0) { wval[tid >> 5] = bv; widx[tid >> 5] = bi; }
        __syncthreads();
        if (tid < 32) {
            bv = wval[tid]; bi = widx[tid];
#pragma unroll
            for (int off = 16; off > 0; off >>= 1) {
                float ov = __shfl_xor(bv, off, 32);
                int   oi = __shfl_xor(bi, off, 32);
                if (ov > bv || (ov == bv && oi < bi)) { bv = ov; bi = oi; }
            }
            if (tid == 0) {
                int keep = (bv > 0.0f) ? 1 : 0;
                bcast_idx  = bi;
                bcast_keep = keep;
                bb[0] = bo[(size_t)bi * 4 + 0];
                bb[1] = bo[(size_t)bi * 4 + 1];
                bb[2] = bo[(size_t)bi * 4 + 2];
                bb[3] = bo[(size_t)bi * 4 + 3];
                idx_k [b * TOPK + t] = bi;
                keep_k[b * TOPK + t] = keep;
            }
        }
        __syncthreads();
        const int   kidx = bcast_idx;
        const int   keep = bcast_keep;
        const float q0 = bb[0], q1 = bb[1], q2 = bb[2], q3 = bb[3];
        const float areaQ = (q2 - q0) * (q3 - q1);
#pragma unroll
        for (int i = 0; i < PER_TH; ++i) {
            int a = tid + i * 1024;
            if (a == kidx) { liv[i] = NEGV; continue; }   // live[idx] = NEG
            if (keep && liv[i] > 0.0f) {
                float p0 = bo[(size_t)a * 4 + 0], p1 = bo[(size_t)a * 4 + 1];
                float p2 = bo[(size_t)a * 4 + 2], p3 = bo[(size_t)a * 4 + 3];
                float ix = fminf(q2, p2) - fmaxf(q0, p0);
                float iy = fminf(q3, p3) - fmaxf(q1, p1);
                float inter = fmaxf(ix, 0.0f) * fmaxf(iy, 0.0f);
                float uni = areaQ + (p2 - p0) * (p3 - p1) - inter;
                float iou = (uni > 0.0f) ? inter / uni : 0.0f;
                if (iou > NMS_T) liv[i] = NEGV;
            }
        }
        // no extra barrier needed: next iteration's shared writes are fenced by
        // the two barriers above before any cross-wave reuse
    }
}

// ---------------- per-class slot via WMMA triangular matmul + scatter -------
// slot[i] = (tril_ones(200x200) @ onehot_kept(200x80))[i][cls_i] - 1
// exact in f16 x f16 -> f32 (integer counts <= 200). 1 wave per image.
__global__ void __launch_bounds__(32) k_slot_scatter(const float* __restrict__ scores,
                                                     const float* __restrict__ boxes,
                                                     const int*   __restrict__ cls,
                                                     const int*   __restrict__ idx_k,
                                                     const int*   __restrict__ keep_k,
                                                     float* __restrict__ out) {
    const int b    = blockIdx.x;
    const int lane = threadIdx.x;
    __shared__ int   tag[TOPK_PAD];          // kept ? class : -1
    __shared__ float S[16][NUM_CLASSES];     // one M-tile of the cumsum matrix

    for (int t = lane; t < TOPK_PAD; t += 32) {
        int tg = -1;
        if (t < TOPK && keep_k[b * TOPK + t]) {
            int id = idx_k[b * TOPK + t];
            tg = cls[(size_t)b * A_PAD + id];
        }
        tag[t] = tg;
    }
    __syncthreads();

    const int half = lane >> 4;      // A/B half-wave selector (ISA VGPR layout)
    const int ml   = lane & 15;

    for (int mt = 0; mt < 13; ++mt) {            // 13 * 16 = 208 >= 200 rows
        v8f acc[5] = {};
        const int ig = mt * 16 + ml;             // global row i for this lane
        for (int kt = 0; kt < 7; ++kt) {         // 7 * 32 = 224 >= 200 K
            const int k0 = kt * 32;
            // A tile: T[i][j] = (j <= i), 16x32 f16, ISA A-layout
            v16h aT;
#pragma unroll
            for (int e = 0; e < 16; ++e) {
                int koff = (e < 8) ? (half * 8 + e) : (16 + half * 8 + (e - 8));
                int j = k0 + koff;
                aT[e] = (j <= ig) ? (_Float16)1.0f : (_Float16)0.0f;
            }
#pragma unroll
            for (int nt = 0; nt < 5; ++nt) {     // 5 * 16 = 80 classes
                const int c = nt * 16 + ml;
                // B tile: H[j][c] = (kept_j && cls_j == c), 32x16 f16, ISA B-layout
                v16h bH;
#pragma unroll
                for (int e = 0; e < 16; ++e) {
                    int j = k0 + ((e < 8) ? (half * 8 + e) : (16 + half * 8 + (e - 8)));
                    bH[e] = (tag[j] == c) ? (_Float16)1.0f : (_Float16)0.0f;
                }
                acc[nt] = __builtin_amdgcn_wmma_f32_16x16x32_f16(
                    false, aT, false, bH, (short)0, acc[nt], false, false);
            }
        }
        // C/D layout: vgpr r, lane -> (M = half*8 + r, N = ml)
#pragma unroll
        for (int nt = 0; nt < 5; ++nt)
#pragma unroll
            for (int r = 0; r < 8; ++r)
                S[half * 8 + r][nt * 16 + ml] = acc[nt][r];
        __syncthreads();

        if (lane < 16) {
            int i = mt * 16 + lane;
            if (i < TOPK && tag[i] >= 0) {
                int c    = tag[i];
                int slot = (int)(S[lane][c] + 0.5f) - 1;   // inclusive count - 1
                int id   = idx_k[b * TOPK + i];
                float sc = scores[(size_t)b * A_PAD + id];
                const float* bp = boxes + ((size_t)b * A_PAD + id) * 4;
                float* op = out + ((((size_t)b * (NUM_CLASSES + 1)) + (c + 1)) * TOPK + slot) * 5;
                op[0] = sc; op[1] = bp[0]; op[2] = bp[1]; op[3] = bp[2]; op[4] = bp[3];
            }
        }
        __syncthreads();
    }
}

// ---------------------------------------------------------------------------
extern "C" void kernel_launch(void* const* d_in, const int* in_sizes, int n_in,
                              void* d_out, int out_size, void* d_ws, size_t ws_size,
                              hipStream_t stream) {
    (void)in_sizes; (void)n_in; (void)ws_size;
    const float* conf = (const float*)d_in[0];   // [B, A, 81]
    const float* reg  = (const float*)d_in[1];   // [B, A, 4]
    const float* anch = (const float*)d_in[2];   // [A, 4]
    float* out = (float*)d_out;                  // [B, 81, 200, 5]

    char* p = (char*)d_ws;
    float* scores  = (float*)p; p += sizeof(float) * BATCH * A_PAD;
    float* boxes   = (float*)p; p += sizeof(float) * BATCH * A_PAD * 4;
    float* boxes_o = (float*)p; p += sizeof(float) * BATCH * A_PAD * 4;
    int*   cls     = (int*)  p; p += sizeof(int)   * BATCH * A_PAD;
    float* maxbox  = (float*)p; p += sizeof(float) * BATCH;
    int*   idx_k   = (int*)  p; p += sizeof(int)   * BATCH * TOPK;
    int*   keep_k  = (int*)  p; p += sizeof(int)   * BATCH * TOPK;

    k_init<<<(out_size + 255) / 256, 256, 0, stream>>>(out, out_size, maxbox);
    dim3 g1(A_PAD / 256, BATCH);
    k_decode<<<g1, 256, 0, stream>>>(conf, reg, anch, scores, boxes, cls, maxbox);
    k_offset<<<g1, 256, 0, stream>>>(boxes, cls, maxbox, boxes_o);
    k_nms<<<BATCH, 1024, 0, stream>>>(scores, boxes_o, idx_k, keep_k);
    k_slot_scatter<<<BATCH, 32, 0, stream>>>(scores, boxes, cls, idx_k, keep_k, out);
}
